// LSTMDynamics_30142080483445
// MI455X (gfx1250) — compile-verified, matrix operand order checked by reading, not run on hardware
//
#include <hip/hip_runtime.h>
#include <hip/hip_bf16.h>

// ---------------------------------------------------------------------------
// 2-layer LSTM (B=64, T=512, D_IN=64, H=512) + projection to D_OUT=32.
// Persistent per-layer kernels, 32 WGs x 512 threads; each WG owns a 16-wide
// hidden slice (weight slice as bf16 in LDS), per-step concat-GEMM
// [x_t | h_{t-1}] @ Wcat^T via v_wmma_f32_16x16x32_bf16, h broadcast through a
// double-buffered global buffer + device-wide atomic barrier per step.
// Elementwise cell update uses branch-free v_exp/v_rcp fast activations to
// keep the 1024-step sequential critical path short.
// ---------------------------------------------------------------------------

#define TSTEPS  512
#define BATCH   64
#define HID     512
#define NWG     32
#define COLS    16     // hidden units per workgroup
#define THREADS 512    // 16 waves (wave32)

typedef __attribute__((ext_vector_type(16))) __bf16 v16bf;
typedef __attribute__((ext_vector_type(8)))  __bf16 v8bf;
typedef __attribute__((ext_vector_type(8)))  float  v8f;

union FragU { v16bf v; struct { v8bf lo; v8bf hi; } p; };

// bf16 A/B fragment for 16x16x32: per lane, elements 0..7 = K k..k+7,
// elements 8..15 = K k+16..k+23, where k = kk*32 + (lane>>4)*8.
// rowptr points at the start of this lane's (M- or N-) row, row-major over K.
__device__ __forceinline__ v16bf ld_frag_bf16(const __bf16* rowptr, int k) {
  FragU f;
  f.p.lo = *(const v8bf*)(rowptr + k);
  f.p.hi = *(const v8bf*)(rowptr + k + 16);
  return f.v;
}

// Branch-free fast activations: v_exp_f32 + v_rcp_f32.
// exp overflow/underflow saturates these correctly at the asymptotes.
__device__ __forceinline__ float fast_sigmoid(float x) {
  return __builtin_amdgcn_rcpf(1.0f + __expf(-x));
}
__device__ __forceinline__ float fast_tanh(float x) {
  return 2.0f * __builtin_amdgcn_rcpf(1.0f + __expf(-2.0f * x)) - 1.0f;
}

// ---------------------------------------------------------------------------
// Persistent LSTM layer kernel. KIN = input feature size (64 or 512).
// Each WG computes gates[:, 4 x 16-col slice] every step with WMMA, then the
// elementwise cell update for its 16 hidden units, then global h-broadcast.
// ---------------------------------------------------------------------------
template<int KIN, bool IN_IS_F32>
__global__ void lstm_layer_kernel(const float*  __restrict__ xin_f32,   // [B,T,KIN] (layer 0)
                                  const __bf16* __restrict__ xin_bf16,  // [T,B,KIN] (layer 1)
                                  const float*  __restrict__ W_ih,      // [4H,KIN]
                                  const float*  __restrict__ W_hh,      // [4H,H]
                                  const float*  __restrict__ b_ih,      // [4H]
                                  const float*  __restrict__ b_hh,      // [4H]
                                  __bf16*       __restrict__ hbuf0,     // [B,H] broadcast, parity 0
                                  __bf16*       __restrict__ hbuf1,     // [B,H] broadcast, parity 1
                                  __bf16*       __restrict__ seq_out,   // [T,B,H] or null
                                  float*        __restrict__ hlast_out, // [B,H] or null
                                  unsigned long long* __restrict__ counter)
{
  constexpr int K = KIN + HID;   // concat GEMM depth
  extern __shared__ char smem[];
  __bf16* Wc  = (__bf16*)smem;                       // [64][K]  weight slice (4 gates x 16 cols)
  __bf16* Ain = (__bf16*)(smem + (size_t)64*K*2);    // [64][K]  concat input [x_t | h_prev]
  float*  gat = (float*) (smem + (size_t)2*64*K*2);  // [4][64][16] gate tiles
  float*  bia = gat + 4*64*16;                       // [64]  combined bias
  float*  cst = bia + 64;                            // [64][16] cell state (persistent)

  const int tid   = threadIdx.x;
  const int wg    = blockIdx.x;
  const int c0    = wg * COLS;
  const int lane  = tid & 31;
  const int w     = tid >> 5;
  const int Mtile = w >> 2;       // batch tile 0..3
  const int gate  = w & 3;        // i,f,g,o
  const int kofs  = (lane >> 4) << 3;

  // ---- one-time setup: stage weight slice (f32 -> bf16), bias, zero c ----
  for (int idx = tid; idx < 64*K; idx += THREADS) {
    int r = idx / K, k = idx - r*K;
    int g = r >> 4, n = r & 15;
    int wrow = g*HID + c0 + n;
    float v = (k < KIN) ? W_ih[wrow*KIN + k] : W_hh[wrow*HID + (k - KIN)];
    Wc[idx] = (__bf16)v;
  }
  if (tid < 64) {
    int g = tid >> 4, n = tid & 15;
    int wrow = g*HID + c0 + n;
    bia[tid] = b_ih[wrow] + b_hh[wrow];
  }
  for (int idx = tid; idx < 64*COLS; idx += THREADS) cst[idx] = 0.0f;
  __syncthreads();

  const __bf16* Arow = Ain + (Mtile*16 + (lane & 15)) * K;
  const __bf16* Brow = Wc  + (gate *16 + (lane & 15)) * K;

  for (int t = 0; t < TSTEPS; ++t) {
    // ---- stage concat input [x_t | h_{t-1}] into LDS as bf16 ----
    for (int idx = tid; idx < 64*KIN; idx += THREADS) {
      int m = idx / KIN, k = idx - m*KIN;
      __bf16 v;
      if constexpr (IN_IS_F32) v = (__bf16)xin_f32[(m*TSTEPS + t)*KIN + k];
      else                     v = xin_bf16[((size_t)t*BATCH + m)*KIN + k];
      Ain[m*K + k] = v;
    }
    const __bf16* hprev = (t & 1) ? hbuf0 : hbuf1;
    for (int idx = tid; idx < 64*HID; idx += THREADS) {
      int m = idx >> 9, k = idx & (HID - 1);
      Ain[m*K + KIN + k] = (t == 0) ? (__bf16)0.0f : hprev[idx];
    }
    __syncthreads();

    // ---- gates tile (Mtile, gate) via chained WMMA over K ----
    v8f acc = {};
    #pragma unroll 4
    for (int kk = 0; kk < K/32; ++kk) {
      v16bf a = ld_frag_bf16(Arow, kk*32 + kofs);
      v16bf b = ld_frag_bf16(Brow, kk*32 + kofs);
      acc = __builtin_amdgcn_wmma_f32_16x16x32_bf16(false, a, false, b,
                                                    (short)0, acc, false, false);
    }
    {
      int n  = lane & 15;
      int mb = Mtile*16 + ((lane >> 4) << 3);
      #pragma unroll
      for (int r = 0; r < 8; ++r)
        gat[(gate*64 + mb + r)*16 + n] = acc[r];
    }
    __syncthreads();

    // ---- elementwise LSTM cell update for this WG's 16 hidden units ----
    __bf16* hw = (t & 1) ? hbuf1 : hbuf0;
    for (int c = tid; c < 64*COLS; c += THREADS) {
      int m = c >> 4, n = c & 15;
      float gi = gat[(0*64 + m)*16 + n] + bia[ 0 + n];
      float gf = gat[(1*64 + m)*16 + n] + bia[16 + n];
      float gg = gat[(2*64 + m)*16 + n] + bia[32 + n];
      float go = gat[(3*64 + m)*16 + n] + bia[48 + n];
      float ii = fast_sigmoid(gi);
      float ff = fast_sigmoid(gf);
      float tg = fast_tanh(gg);
      float oo = fast_sigmoid(go);
      float cp = ff * cst[c] + ii * tg;
      float hp = oo * fast_tanh(cp);
      cst[c] = cp;
      __bf16 hb = (__bf16)hp;
      hw[m*HID + c0 + n] = hb;
      if (seq_out)
        seq_out[((size_t)t*BATCH + m)*HID + c0 + n] = hb;
      if (hlast_out && t == TSTEPS-1)
        hlast_out[m*HID + c0 + n] = hp;
    }

    // ---- device-wide barrier (monotonic counter; skip after last step) ----
    if (t < TSTEPS-1) {
      __syncthreads();
      if (tid == 0) {
        __threadfence();
        atomicAdd(counter, 1ull);
        unsigned long long target = (unsigned long long)(t + 1) * NWG;
        while (__hip_atomic_load(counter, __ATOMIC_RELAXED,
                                 __HIP_MEMORY_SCOPE_AGENT) < target)
          __builtin_amdgcn_s_sleep(1);
      }
      __syncthreads();
      __threadfence();
    }
  }
}

// ---------------------------------------------------------------------------
// Final projection: out[64,32] = h_last[64,512] @ W_out[32,512]^T + b_out.
// One block, 8 waves, 8 WMMA output tiles (4 M-tiles x 2 N-tiles), K=512.
// ---------------------------------------------------------------------------
__global__ void out_proj_kernel(const float* __restrict__ hlast,
                                const float* __restrict__ W_out,
                                const float* __restrict__ b_out,
                                float*       __restrict__ out)
{
  extern __shared__ char smem[];
  __bf16* A  = (__bf16*)smem;                       // [64][512]
  __bf16* Bw = (__bf16*)(smem + (size_t)64*512*2);  // [32][512]
  int tid = threadIdx.x;
  for (int idx = tid; idx < 64*512; idx += 256) A[idx]  = (__bf16)hlast[idx];
  for (int idx = tid; idx < 32*512; idx += 256) Bw[idx] = (__bf16)W_out[idx];
  __syncthreads();

  int lane = tid & 31, w = tid >> 5;
  int Mtile = w >> 1, Ntile = w & 1;
  int kofs = (lane >> 4) << 3;
  const __bf16* Arow = A  + (Mtile*16 + (lane & 15)) * 512;
  const __bf16* Brow = Bw + (Ntile*16 + (lane & 15)) * 512;

  v8f acc = {};
  #pragma unroll
  for (int kk = 0; kk < 16; ++kk) {
    v16bf a = ld_frag_bf16(Arow, kk*32 + kofs);
    v16bf b = ld_frag_bf16(Brow, kk*32 + kofs);
    acc = __builtin_amdgcn_wmma_f32_16x16x32_bf16(false, a, false, b,
                                                  (short)0, acc, false, false);
  }
  int n  = lane & 15;
  int mb = Mtile*16 + ((lane >> 4) << 3);
  float bo = b_out[Ntile*16 + n];
  #pragma unroll
  for (int r = 0; r < 8; ++r)
    out[(mb + r)*32 + Ntile*16 + n] = acc[r] + bo;
}

// ---------------------------------------------------------------------------
extern "C" void kernel_launch(void* const* d_in, const int* in_sizes, int n_in,
                              void* d_out, int out_size, void* d_ws, size_t ws_size,
                              hipStream_t stream) {
  const float* x     = (const float*)d_in[0];
  const float* W_ih0 = (const float*)d_in[1];
  const float* W_hh0 = (const float*)d_in[2];
  const float* b_ih0 = (const float*)d_in[3];
  const float* b_hh0 = (const float*)d_in[4];
  const float* W_ih1 = (const float*)d_in[5];
  const float* W_hh1 = (const float*)d_in[6];
  const float* b_ih1 = (const float*)d_in[7];
  const float* b_hh1 = (const float*)d_in[8];
  const float* W_out = (const float*)d_in[9];
  const float* b_out = (const float*)d_in[10];

  // Workspace layout:
  //   [0,256)        : two 8B barrier counters (zeroed each call)
  //   [256, +128KB)  : h broadcast double buffer, 2 x [64,512] bf16
  //   next 32MB      : layer-0 output sequence [T,64,512] bf16
  //   next 128KB     : h_last [64,512] f32
  char* ws = (char*)d_ws;
  unsigned long long* ctr0 = (unsigned long long*)ws;
  unsigned long long* ctr1 = ctr0 + 2;
  __bf16* hbuf0 = (__bf16*)(ws + 256);
  __bf16* hbuf1 = (__bf16*)(ws + 256 + (size_t)BATCH*HID*2);
  __bf16* h1seq = (__bf16*)(ws + 256 + (size_t)2*BATCH*HID*2);
  float*  hlast = (float*) (ws + 256 + (size_t)2*BATCH*HID*2
                               + (size_t)TSTEPS*BATCH*HID*2);

  hipMemsetAsync(d_ws, 0, 256, stream);  // reset barrier counters (graph-safe)

  const size_t smem0 = (size_t)2*64*(64  + HID)*2 + 4*64*16*4 + 64*4 + 64*16*4; // ~168KB
  const size_t smem1 = (size_t)2*64*(512 + HID)*2 + 4*64*16*4 + 64*4 + 64*16*4; // ~283KB
  const size_t smem2 = (size_t)64*512*2 + (size_t)32*512*2;                     //  96KB
  hipFuncSetAttribute(reinterpret_cast<const void*>(&lstm_layer_kernel<64, true>),
                      hipFuncAttributeMaxDynamicSharedMemorySize, (int)smem0);
  hipFuncSetAttribute(reinterpret_cast<const void*>(&lstm_layer_kernel<512, false>),
                      hipFuncAttributeMaxDynamicSharedMemorySize, (int)smem1);
  hipFuncSetAttribute(reinterpret_cast<const void*>(&out_proj_kernel),
                      hipFuncAttributeMaxDynamicSharedMemorySize, (int)smem2);

  lstm_layer_kernel<64, true><<<NWG, THREADS, smem0, stream>>>(
      x, nullptr, W_ih0, W_hh0, b_ih0, b_hh0,
      hbuf0, hbuf1, h1seq, nullptr, ctr0);

  lstm_layer_kernel<512, false><<<NWG, THREADS, smem1, stream>>>(
      nullptr, h1seq, W_ih1, W_hh1, b_ih1, b_hh1,
      hbuf0, hbuf1, nullptr, hlast, ctr1);

  out_proj_kernel<<<1, 256, smem2, stream>>>(hlast, W_out, b_out, (float*)d_out);
}
